// Efficient_DiceScore_6674379178263
// MI455X (gfx1250) — compile-verified, matrix operand order checked by reading, not run on hardware
//
#include <hip/hip_runtime.h>
#include <hip/hip_bf16.h>

typedef __attribute__((ext_vector_type(4)))  float    v4f;
typedef __attribute__((ext_vector_type(4)))  int      v4i;
typedef __attribute__((ext_vector_type(16))) _Float16 v16h;
typedef __attribute__((ext_vector_type(8)))  float    v8f;

#define NCLS   5
#define PLANE  65536            // 256*256 floats per (z, class) slice
#define NVOX   (256 * 65536)    // 16,777,216 voxels
#define BLOCKS 2048
#define TPB    256              // 8 wave32 per block

__global__ void dice_zero(unsigned* __restrict__ g) {
    if (threadIdx.x < 16) g[threadIdx.x] = 0u;
}

__global__ __launch_bounds__(TPB) void dice_count(const float* __restrict__ in,
                                                  const int*   __restrict__ tgt,
                                                  unsigned*    __restrict__ g) {
    // per-wave staging for the WMMA B matrix: [wave][K=lane][N=counter]
    __shared__ _Float16 stage[8][32][16];
    __shared__ unsigned blk[16];

    const int tid  = blockIdx.x * TPB + threadIdx.x;
    const int lane = threadIdx.x & 31;
    const int wave = threadIdx.x >> 5;
    if (threadIdx.x < 16) blk[threadIdx.x] = 0u;

    // 0..4 pred counts, 5..9 target counts, 10..14 intersection counts
    unsigned cnt[15];
#pragma unroll
    for (int i = 0; i < 15; ++i) cnt[i] = 0u;

    const int nv4    = NVOX / 4;        // 4,194,304 float4 voxel groups
    const int stride = BLOCKS * TPB;    // 524,288 -> exactly 8 iterations/thread
    for (int i = tid; i < nv4; i += stride) {
        const int v   = i << 2;
        const int z   = v >> 16;        // slice
        const int rem = v & 65535;      // offset inside the 256x256 plane (16B aligned)
        const float* base = in + (size_t)(z * NCLS) * PLANE + rem;
        // streaming reads: NT hint (no reuse; 400MB stream > 192MB L2)
        v4f c0 = __builtin_nontemporal_load((const v4f*)(base + 0 * PLANE));
        v4f c1 = __builtin_nontemporal_load((const v4f*)(base + 1 * PLANE));
        v4f c2 = __builtin_nontemporal_load((const v4f*)(base + 2 * PLANE));
        v4f c3 = __builtin_nontemporal_load((const v4f*)(base + 3 * PLANE));
        v4f c4 = __builtin_nontemporal_load((const v4f*)(base + 4 * PLANE));
        v4i t  = __builtin_nontemporal_load((const v4i*)(tgt + v));
#pragma unroll
        for (int e = 0; e < 4; ++e) {
            float a0 = c0[e], a1 = c1[e], a2 = c2[e], a3 = c3[e], a4 = c4[e];
            // first-max tie-breaking like jnp.argmax: strict '>'
            int best = 0; float bv = a0;
            if (a1 > bv) { bv = a1; best = 1; }
            if (a2 > bv) { bv = a2; best = 2; }
            if (a3 > bv) { bv = a3; best = 3; }
            if (a4 > bv) { bv = a4; best = 4; }
            const int tv = t[e];
#pragma unroll
            for (int c = 0; c < 5; ++c) {
                unsigned p = (best == c) ? 1u : 0u;
                unsigned q = (tv  == c) ? 1u : 0u;
                cnt[c]      += p;
                cnt[5 + c]  += q;
                cnt[10 + c] += (p & q);
            }
        }
    }

    // ---- wave-level reduction via v_wmma_f32_16x16x32_f16 ----
    // D = ones(16x32) * B(32x16), B[k][n] = counter n of lane k.
    // Counters <= 32 voxels/thread -> exact in f16; f32 accumulate exact.
#pragma unroll
    for (int c = 0; c < 15; ++c) stage[wave][lane][c] = (_Float16)cnt[c];
    stage[wave][lane][15] = (_Float16)0.0f;
    __syncthreads();

    const int col   = lane & 15;            // N index this lane supplies
    const int kbase = (lane >> 4) << 4;     // lanes 0-15: K=0..15, lanes 16-31: K=16..31
    v16h b;
#pragma unroll
    for (int j = 0; j < 8; ++j) {           // VGPR j holds rows K=2j (lo) and 2j+1 (hi)
        b[2 * j]     = stage[wave][kbase + 2 * j][col];
        b[2 * j + 1] = stage[wave][kbase + 2 * j + 1][col];
    }
    v16h a;
#pragma unroll
    for (int j = 0; j < 16; ++j) a[j] = (_Float16)1.0f;
    v8f acc = {};
    acc = __builtin_amdgcn_wmma_f32_16x16x32_f16(
        /*neg_a=*/false, a, /*neg_b=*/false, b,
        /*c_mod=*/(short)0, acc, /*reuse_a=*/false, /*reuse_b=*/false);

    // D row M=0 (VGPR0, lanes 0-15): wave-total of counter 'lane'
    const unsigned tot = (unsigned)(acc[0] + 0.5f);
    if (lane < 15) atomicAdd(&blk[lane], tot);      // 8 waves -> block totals (LDS, int)
    __syncthreads();
    if (threadIdx.x < 15) atomicAdd(&g[threadIdx.x], blk[threadIdx.x]); // 15 atomics/block
}

__global__ void dice_final(const unsigned* __restrict__ g, float* __restrict__ out) {
    const int c = threadIdx.x;
    if (c < 4) {  // ignore_chan0: classes 1..4
        const float inter = (float)g[10 + 1 + c];
        const float np    = (float)g[0  + 1 + c];
        const float ng    = (float)g[5  + 1 + c];
        out[c] = 2.0f * inter / (np + ng + 1e-7f);
    }
}

extern "C" void kernel_launch(void* const* d_in, const int* in_sizes, int n_in,
                              void* d_out, int out_size, void* d_ws, size_t ws_size,
                              hipStream_t stream) {
    const float* in  = (const float*)d_in[0];   // [256,5,256,256] f32
    const int*   tgt = (const int*)d_in[1];     // [256,1,256,256] i32
    unsigned*    g   = (unsigned*)d_ws;         // 15 global counters
    float*       out = (float*)d_out;           // 4 dice scores

    dice_zero <<<1, 32, 0, stream>>>(g);
    dice_count<<<BLOCKS, TPB, 0, stream>>>(in, tgt, g);
    dice_final<<<1, 32, 0, stream>>>(g, out);
}
